// AlignOnlyModel_55645596287315
// MI455X (gfx1250) — matrix operand order repaired and checked by reading, not compiled, not run on hardware
//
#include <hip/hip_runtime.h>

typedef __attribute__((ext_vector_type(16))) __bf16         v16bf;
typedef __attribute__((ext_vector_type(8)))  float          v8f;
typedef __attribute__((ext_vector_type(4)))  unsigned int   u32x4;
typedef __attribute__((ext_vector_type(4)))  float          f32x4;
typedef __attribute__((ext_vector_type(4)))  unsigned short u16x4;

#define N_NODES 50000
#define N_EDGES 800000
#define LM_DIM  768
#define ALIGN_D 256
#define BATCH   1024
#define SEQ_L   128
#define B_SEL   768

// ---------- helpers ----------
__device__ __forceinline__ unsigned short f2bf(float f) {
  unsigned int u = __float_as_uint(f);
  u += 0x7FFFu + ((u >> 16) & 1u);          // round-to-nearest-even
  return (unsigned short)(u >> 16);
}

// hardware f32 atomic add (avoid CAS-loop lowering)
__device__ __forceinline__ void gatomic_add_f32(float* addr, float val) {
  asm volatile("global_atomic_add_f32 %0, %1, off" : : "v"(addr), "v"(val) : "memory");
}

// ---------- elementwise kernels (float4-vectorized) ----------
__global__ void zero_kernel(f32x4* p, int n4) {
  int i = blockIdx.x * blockDim.x + threadIdx.x;
  if (i < n4) p[i] = (f32x4){0.f, 0.f, 0.f, 0.f};
}

__global__ void convert_bf16_kernel(const f32x4* __restrict__ in,
                                    u16x4* __restrict__ out, int n4) {
  int i = blockIdx.x * blockDim.x + threadIdx.x;
  if (i < n4) {
    f32x4 v = in[i];
    u16x4 o;
    o.x = f2bf(v.x); o.y = f2bf(v.y); o.z = f2bf(v.z); o.w = f2bf(v.w);
    out[i] = o;
  }
}

// weights: [K,N] f32 -> [N,K] bf16 (column-major) so B fragments are contiguous
__global__ void convert_transpose_bf16_kernel(const float* __restrict__ in,
                                              unsigned short* __restrict__ out,
                                              int K, int N) {
  int idx = blockIdx.x * blockDim.x + threadIdx.x;
  if (idx >= K * N) return;
  int n = idx / K, k = idx - n * K;
  out[idx] = f2bf(in[(size_t)k * N + n]);
}

// mean-pool over tokens, emit bf16 pooled activations [B, LM_DIM]
__global__ void pool_kernel(const float* __restrict__ x,
                            unsigned short* __restrict__ out) {
  int b = blockIdx.x;
  const float* base = x + (size_t)b * SEQ_L * LM_DIM;
  for (int d = threadIdx.x; d < LM_DIM; d += blockDim.x) {
    float s = 0.0f;
    #pragma unroll 4
    for (int t = 0; t < SEQ_L; ++t) s += base[t * LM_DIM + d];
    out[(size_t)b * LM_DIM + d] = f2bf(s * (1.0f / (float)SEQ_L));
  }
}

// ---------- graph structure ----------
__global__ void deg_kernel(const int* __restrict__ dst, float* deg, int E) {
  int e = blockIdx.x * blockDim.x + threadIdx.x;
  if (e < E) gatomic_add_f32(&deg[dst[e]], 1.0f);
}

__global__ void dinv_kernel(const float* __restrict__ deg, float* dinv, int n) {
  int i = blockIdx.x * blockDim.x + threadIdx.x;
  if (i < n) dinv[i] = rsqrtf(deg[i] + 1.0f);   // +1 for self loop
}

// one wave per edge: agg[dst] += h[src] * dinv[src]*dinv[dst]; b128 gathers
__global__ void edge_agg_kernel(const int* __restrict__ src, const int* __restrict__ dst,
                                const float* __restrict__ h, const float* __restrict__ dinv,
                                float* __restrict__ agg, int E, int dim) {
  int gid  = blockIdx.x * blockDim.x + threadIdx.x;
  int e    = gid >> 5;
  int lane = gid & 31;
  if (e >= E) return;
  int s = src[e], t = dst[e];
  float norm = dinv[s] * dinv[t];
  const f32x4* hs = (const f32x4*)(h + (size_t)s * dim);
  float*       at = agg + (size_t)t * dim;
  int dim4 = dim >> 2;
  for (int d = lane; d < dim4; d += 32) {
    f32x4 v = hs[d];
    int b = d << 2;
    gatomic_add_f32(at + b + 0, v.x * norm);
    gatomic_add_f32(at + b + 1, v.y * norm);
    gatomic_add_f32(at + b + 2, v.z * norm);
    gatomic_add_f32(at + b + 3, v.w * norm);
  }
}

// agg + h*dinv^2 + bias, LeakyReLU, emit bf16 for next layer's GEMM (x4 vector)
__global__ void post1_kernel(const f32x4* __restrict__ agg, const f32x4* __restrict__ h,
                             const float* __restrict__ dinv, const float* __restrict__ b,
                             u16x4* __restrict__ out, int dim) {
  int i = blockIdx.x * blockDim.x + threadIdx.x;       // over N_NODES*dim/4
  int base = i << 2;
  int node = base / dim, d = base - node * dim;
  float di = dinv[node];
  float dd = di * di;
  f32x4 a = agg[i], hv = h[i];
  float v0 = a.x + hv.x * dd + b[d + 0];
  float v1 = a.y + hv.y * dd + b[d + 1];
  float v2 = a.z + hv.z * dd + b[d + 2];
  float v3 = a.w + hv.w * dd + b[d + 3];
  v0 = v0 > 0.f ? v0 : 0.01f * v0;
  v1 = v1 > 0.f ? v1 : 0.01f * v1;
  v2 = v2 > 0.f ? v2 : 0.01f * v2;
  v3 = v3 > 0.f ? v3 : 0.01f * v3;
  u16x4 o; o.x = f2bf(v0); o.y = f2bf(v1); o.z = f2bf(v2); o.w = f2bf(v3);
  out[i] = o;
}

// output gather: fuse conv2 self-loop + bias for the 768 selected rows only
__global__ void gather_kernel(const float* __restrict__ agg, const float* __restrict__ h,
                              const float* __restrict__ dinv, const float* __restrict__ b,
                              const int* __restrict__ batch_idx, const int* __restrict__ mask,
                              float* __restrict__ out) {
  int i = blockIdx.x, c = threadIdx.x;                 // 768 x 256
  int node = batch_idx[mask[i]];
  float di = dinv[node];
  size_t off = (size_t)node * ALIGN_D + c;
  out[(size_t)i * ALIGN_D + c] = agg[off] + h[off] * di * di + b[c];
}

// ---------- WMMA bf16 GEMM: C[M,N] = A[M,K] @ Bt[N,K]^T (+bias) ----------
// 8 waves/block, each wave a 16x16 tile; block tile 32x64. M % 16 == 0.
union Frag { u32x4 u[2]; v16bf v; };

__global__ __launch_bounds__(256)
void wmma_gemm_bf16(const unsigned short* __restrict__ A,   // [M,K] bf16 row-major
                    const unsigned short* __restrict__ Bt,  // [N,K] bf16 (W transposed)
                    const float* __restrict__ bias,
                    float* __restrict__ C, int M, int N, int K) {
  int lane = threadIdx.x & 31;
  int wave = threadIdx.x >> 5;            // 0..7
  int m0 = blockIdx.x * 32 + (wave >> 2) * 16;
  int n0 = blockIdx.y * 64 + (wave & 3) * 16;
  int hh = lane >> 4;                     // half-wave 0/1
  int r  = lane & 15;

  int arow = m0 + r; if (arow >= M) arow = M - 1;       // clamp loads
  const unsigned short* Ap = A  + (size_t)arow * K + hh * 8;
  const unsigned short* Bp = Bt + (size_t)(n0 + r) * K + hh * 8;

  v8f acc = {};
  #pragma unroll 4
  for (int k = 0; k < K; k += 32) {
    Frag fa, fb;
    // A 16x32: lane holds rows of two contiguous 8-elem runs (16B each)
    fa.u[0] = *(const u32x4*)(Ap + k);
    fa.u[1] = *(const u32x4*)(Ap + k + 16);
    // B 32x16: lane = column, two 8-elem K runs matching A (pre-transposed)
    fb.u[0] = *(const u32x4*)(Bp + k);
    fb.u[1] = *(const u32x4*)(Bp + k + 16);
    __builtin_prefetch(Ap + k + 256, 0, 0);             // global_prefetch_b8
    acc = __builtin_amdgcn_wmma_f32_16x16x32_bf16(
        false, fa.v, false, fb.v, (short)0, acc, false, false);
  }
  if (m0 >= M) return;                                   // uniform tile guard (M%16==0)
  float bv = bias ? bias[n0 + r] : 0.0f;
  float* Cp = C + (size_t)(m0 + hh * 8) * N + (n0 + r);  // one base, 8 offset stores
  #pragma unroll
  for (int j = 0; j < 8; ++j) Cp[(size_t)j * N] = acc[j] + bv;
}

// ---------- launch ----------
extern "C" void kernel_launch(void* const* d_in, const int* in_sizes, int n_in,
                              void* d_out, int out_size, void* d_ws, size_t ws_size,
                              hipStream_t stream) {
  const float* x_text  = (const float*)d_in[0];   // [1024,128,768]
  const float* x_graph = (const float*)d_in[1];   // [50000,768]
  const float* W_text  = (const float*)d_in[2];   // [768,256]
  const float* b_text  = (const float*)d_in[3];
  const float* W1      = (const float*)d_in[4];   // [768,512]
  const float* b1      = (const float*)d_in[5];
  const float* W2      = (const float*)d_in[6];   // [512,256]
  const float* b2      = (const float*)d_in[7];
  const int*   src     = (const int*)d_in[8];             // edge_index[0]
  const int*   dst     = ((const int*)d_in[8]) + N_EDGES; // edge_index[1]
  const int*   batch_idx = (const int*)d_in[9];
  const int*   data_mask = (const int*)d_in[10];
  float* out = (float*)d_out;

  // workspace layout
  char* ws = (char*)d_ws;
  size_t o = 0;
  float* deg  = (float*)(ws + o); o += (size_t)N_NODES * 4;
  float* dinv = (float*)(ws + o); o += (size_t)N_NODES * 4;
  o = (o + 255) & ~(size_t)255;
  unsigned short* Abf = (unsigned short*)(ws + o); o += (size_t)N_NODES * LM_DIM * 2;  // reused 50000x512
  unsigned short* Wbf = (unsigned short*)(ws + o); o += (size_t)LM_DIM * 512 * 2;      // transposed weights
  unsigned short* Tbf = (unsigned short*)(ws + o); o += (size_t)BATCH * LM_DIM * 2;    // pooled text
  o = (o + 255) & ~(size_t)255;
  float* H   = (float*)(ws + o); o += (size_t)N_NODES * 512 * 4;   // GEMM out (reused)
  float* AGG = (float*)(ws + o); o += (size_t)N_NODES * 512 * 4;   // atomic accumulator (reused)

  const int T = 256;

  // ---- text branch: pool -> bf16 GEMM (writes out[0 : 1024*256]) ----
  pool_kernel<<<BATCH, T, 0, stream>>>(x_text, Tbf);
  convert_transpose_bf16_kernel<<<(LM_DIM * ALIGN_D + T - 1) / T, T, 0, stream>>>(
      W_text, Wbf, LM_DIM, ALIGN_D);
  wmma_gemm_bf16<<<dim3(BATCH / 32, ALIGN_D / 64), T, 0, stream>>>(
      Tbf, Wbf, b_text, out, BATCH, ALIGN_D, LM_DIM);

  // ---- graph structure ----
  zero_kernel<<<(N_NODES / 4 + T - 1) / T, T, 0, stream>>>((f32x4*)deg, N_NODES / 4);
  zero_kernel<<<(N_NODES * 512 / 4 + T - 1) / T, T, 0, stream>>>((f32x4*)AGG, N_NODES * 512 / 4);
  deg_kernel<<<(N_EDGES + T - 1) / T, T, 0, stream>>>(dst, deg, N_EDGES);
  dinv_kernel<<<(N_NODES + T - 1) / T, T, 0, stream>>>(deg, dinv, N_NODES);

  // ---- conv1: H = bf16(x_graph) @ bf16(W1) ; scatter ; fused act -> bf16 ----
  convert_bf16_kernel<<<(N_NODES * LM_DIM / 4 + T - 1) / T, T, 0, stream>>>(
      (const f32x4*)x_graph, (u16x4*)Abf, N_NODES * LM_DIM / 4);
  convert_transpose_bf16_kernel<<<(LM_DIM * 512 + T - 1) / T, T, 0, stream>>>(
      W1, Wbf, LM_DIM, 512);
  wmma_gemm_bf16<<<dim3((N_NODES + 31) / 32, 512 / 64), T, 0, stream>>>(
      Abf, Wbf, nullptr, H, N_NODES, 512, LM_DIM);
  edge_agg_kernel<<<(N_EDGES * 32 + T - 1) / T, T, 0, stream>>>(src, dst, H, dinv, AGG, N_EDGES, 512);
  post1_kernel<<<(N_NODES * 512 / 4 + T - 1) / T, T, 0, stream>>>(
      (const f32x4*)AGG, (const f32x4*)H, dinv, b1, (u16x4*)Abf, 512);

  // ---- conv2: H2 = act_bf16 @ bf16(W2) ; scatter ; fused gather to out ----
  zero_kernel<<<(N_NODES * ALIGN_D / 4 + T - 1) / T, T, 0, stream>>>(
      (f32x4*)AGG, N_NODES * ALIGN_D / 4);
  convert_transpose_bf16_kernel<<<(512 * ALIGN_D + T - 1) / T, T, 0, stream>>>(
      W2, Wbf, 512, ALIGN_D);
  wmma_gemm_bf16<<<dim3((N_NODES + 31) / 32, ALIGN_D / 64), T, 0, stream>>>(
      Abf, Wbf, nullptr, H, N_NODES, ALIGN_D, 512);
  edge_agg_kernel<<<(N_EDGES * 32 + T - 1) / T, T, 0, stream>>>(src, dst, H, dinv, AGG, N_EDGES, ALIGN_D);
  gather_kernel<<<B_SEL, ALIGN_D, 0, stream>>>(AGG, H, dinv, b2, batch_idx, data_mask,
                                               out + (size_t)BATCH * ALIGN_D);
}
